// HeteroGNN_68221260530191
// MI455X (gfx1250) — matrix operand order repaired and checked by reading, not compiled, not run on hardware
//
#include <hip/hip_runtime.h>
#include <hip/hip_bf16.h>
#include <math.h>

typedef __attribute__((ext_vector_type(16))) _Float16 v16h;
typedef __attribute__((ext_vector_type(8)))  _Float16 v8h;
typedef __attribute__((ext_vector_type(8)))  float    v8f;

#define NEG_GAT 0.2f
#define NEG_LR  0.01f

__device__ __forceinline__ float lrelu(float v, float s){ return v > 0.0f ? v : v * s; }
// order-preserving float<->uint encoding for atomic max on floats
__device__ __forceinline__ unsigned fenc(float f){ unsigned u = __float_as_uint(f); return (u & 0x80000000u) ? ~u : (u | 0x80000000u); }
__device__ __forceinline__ float fdec(unsigned u){ return __uint_as_float((u & 0x80000000u) ? (u & 0x7fffffffu) : ~u); }

union Uh { v16h v; v8h h[2]; };

// ---------------- conversions with padding ----------------
// f32 [M,K] row-major -> f16 [Mp,Kp] row-major, zero padded
__global__ void k_a2h(const float* __restrict__ s, _Float16* __restrict__ d,
                      long M, int K, long Mp, int Kp){
  long i = blockIdx.x * (long)blockDim.x + threadIdx.x;
  long st = (long)gridDim.x * blockDim.x;
  long tot = Mp * (long)Kp;
  for (; i < tot; i += st){
    long r = i / Kp; int c = (int)(i % Kp);
    d[i] = (r < M && c < K) ? (_Float16)s[r * (long)K + c] : (_Float16)0.f;
  }
}
// weight f32 [K,N] -> f16 transposed [N,Kp], zero padded (fragment-friendly)
__global__ void k_w2ht(const float* __restrict__ w, _Float16* __restrict__ d,
                       int K, int N, int Kp){
  long i = blockIdx.x * (long)blockDim.x + threadIdx.x;
  long st = (long)gridDim.x * blockDim.x;
  long tot = (long)N * Kp;
  for (; i < tot; i += st){
    int n = (int)(i / Kp), k = (int)(i % Kp);
    d[i] = (k < K) ? (_Float16)w[(long)k * N + n] : (_Float16)0.f;
  }
}
__global__ void k_fill_f32(float* __restrict__ p, float v, long n){
  long i = blockIdx.x * (long)blockDim.x + threadIdx.x;
  long st = (long)gridDim.x * blockDim.x;
  for (; i < n; i += st) p[i] = v;
}
__global__ void k_fill_u32(unsigned* __restrict__ p, unsigned v, long n){
  long i = blockIdx.x * (long)blockDim.x + threadIdx.x;
  long st = (long)gridDim.x * blockDim.x;
  for (; i < n; i += st) p[i] = v;
}

// ---------------- WMMA GEMM (register-direct fragments, no LDS, no branches) ----
// C[M,N] = A[Mp,Kp]f16 @ W[K,N] (W given as Bt = [N,Kp] f16 transposed)
// block = (32,4): 4 waves; wave handles 16 rows x N cols (NT=N/16 tiles).
// A fragment = two b128 loads (K runs 8*half+0..7 and 16+8*half+0..7).
// B fragment = two b128 loads of transposed weight row (K runs 8*half+0..7 and 16+8*half+0..7).
// All fragment loads for a K-step are issued before the WMMA chain so the
// compiler can clause them and overlap WMMA with in-flight loads.
template<int NT>
__global__ void __launch_bounds__(128) k_gemm_wmma(
    const _Float16* __restrict__ A, const _Float16* __restrict__ Bt,
    const float* __restrict__ bias, float* __restrict__ C,
    int M, int Kp, int N, int act, float slope)
{
  const int lane = threadIdx.x;               // 0..31
  const int wv   = threadIdx.y;               // 0..3
  const int m0   = (blockIdx.x * 4 + wv) * 16;
  const int half = lane >> 4, l15 = lane & 15;
  const _Float16* Arow = A + (long)(m0 + l15) * Kp;
  v8f acc[NT];
  #pragma unroll
  for (int nt = 0; nt < NT; ++nt) acc[nt] = (v8f){};
  const int KT = Kp >> 5;
  for (int kt = 0; kt < KT; ++kt){
    const int kb = kt * 32;
    Uh a, b[NT];
    a.h[0] = *(const v8h*)(Arow + kb + 8 * half);
    a.h[1] = *(const v8h*)(Arow + kb + 16 + 8 * half);
    #pragma unroll
    for (int nt = 0; nt < NT; ++nt){
      const _Float16* Brow = Bt + (long)(nt * 16 + l15) * Kp;
      b[nt].h[0] = *(const v8h*)(Brow + kb + 8 * half);
      b[nt].h[1] = *(const v8h*)(Brow + kb + 16 + 8 * half);
    }
    #pragma unroll
    for (int nt = 0; nt < NT; ++nt)
      acc[nt] = __builtin_amdgcn_wmma_f32_16x16x32_f16(false, a.v, false, b[nt].v,
                                                       (short)0, acc[nt], false, false);
  }
  #pragma unroll
  for (int nt = 0; nt < NT; ++nt){
    #pragma unroll
    for (int r = 0; r < 8; ++r){
      int row = m0 + r + 8 * half;
      if (row < M){
        float v = acc[nt][r];
        int col = nt * 16 + l15;
        if (bias) v += bias[col];
        if (act)  v = lrelu(v, slope);
        C[(long)row * N + col] = v;
      }
    }
  }
}

// ---------------- per-node score dot: s[i] = g[i,:] . a ----------------
__global__ void k_dot(const float* __restrict__ g, const float* __restrict__ a,
                      float* __restrict__ s, long n, int C){
  long i = blockIdx.x * (long)blockDim.x + threadIdx.x;
  long st = (long)gridDim.x * blockDim.x;
  for (; i < n; i += st){
    float acc = 0.f;
    for (int c = 0; c < C; ++c) acc += g[i * (long)C + c] * a[c];
    s[i] = acc;
  }
}

// ---------------- edge softmax passes ----------------
__global__ void k_edge_max(const int* __restrict__ src, const int* __restrict__ dst,
                           const float* __restrict__ ss, const float* __restrict__ sd,
                           unsigned* __restrict__ mx, long E){
  long i = blockIdx.x * (long)blockDim.x + threadIdx.x;
  long st = (long)gridDim.x * blockDim.x;
  for (; i < E; i += st){
    int d = dst[i];
    float e = lrelu(ss[src[i]] + sd[d], NEG_GAT);
    atomicMax(&mx[d], fenc(e));
  }
}
__global__ void k_loop_max(const float* __restrict__ ss, const float* __restrict__ sd,
                           unsigned* __restrict__ mx, long n){
  long i = blockIdx.x * (long)blockDim.x + threadIdx.x;
  long st = (long)gridDim.x * blockDim.x;
  for (; i < n; i += st){
    float e = lrelu(ss[i] + sd[i], NEG_GAT);
    atomicMax(&mx[i], fenc(e));
  }
}
__global__ void k_edge_sum(const int* __restrict__ src, const int* __restrict__ dst,
                           const float* __restrict__ ss, const float* __restrict__ sd,
                           const unsigned* __restrict__ mx, float* __restrict__ dn, long E){
  long i = blockIdx.x * (long)blockDim.x + threadIdx.x;
  long st = (long)gridDim.x * blockDim.x;
  for (; i < E; i += st){
    int d = dst[i];
    float e = lrelu(ss[src[i]] + sd[d], NEG_GAT);
    atomicAdd(&dn[d], __expf(e - fdec(mx[d])));
  }
}
__global__ void k_loop_sum(const float* __restrict__ ss, const float* __restrict__ sd,
                           const unsigned* __restrict__ mx, float* __restrict__ dn, long n){
  long i = blockIdx.x * (long)blockDim.x + threadIdx.x;
  long st = (long)gridDim.x * blockDim.x;
  for (; i < n; i += st){
    float e = lrelu(ss[i] + sd[i], NEG_GAT);
    atomicAdd(&dn[i], __expf(e - fdec(mx[i])));
  }
}
// warp-per-edge weighted scatter: out[dst,:] += alpha * gs[src,:]
__global__ void k_edge_scatter(const int* __restrict__ src, const int* __restrict__ dst,
                               const float* __restrict__ ss, const float* __restrict__ sd,
                               const unsigned* __restrict__ mx, const float* __restrict__ dn,
                               const float* __restrict__ gs, float* __restrict__ out,
                               long E, int C){
  long w  = (blockIdx.x * (long)blockDim.x + threadIdx.x) >> 5;
  long nw = ((long)gridDim.x * blockDim.x) >> 5;
  int lane = threadIdx.x & 31;
  for (long e = w; e < E; e += nw){
    int s = src[e], d = dst[e];
    __builtin_prefetch(&gs[(long)s * C], 0, 1);
    float lg = lrelu(ss[s] + sd[d], NEG_GAT);
    float al = __expf(lg - fdec(mx[d])) / dn[d];
    for (int c = lane; c < C; c += 32)
      atomicAdd(&out[(long)d * C + c], gs[(long)s * C + c] * al);
  }
}
__global__ void k_loop_scatter(const float* __restrict__ ss, const float* __restrict__ sd,
                               const unsigned* __restrict__ mx, const float* __restrict__ dn,
                               const float* __restrict__ gs, float* __restrict__ out,
                               long n, int C){
  long w  = (blockIdx.x * (long)blockDim.x + threadIdx.x) >> 5;
  long nw = ((long)gridDim.x * blockDim.x) >> 5;
  int lane = threadIdx.x & 31;
  for (long i = w; i < n; i += nw){
    float lg = lrelu(ss[i] + sd[i], NEG_GAT);
    float al = __expf(lg - fdec(mx[i])) / dn[i];
    for (int c = lane; c < C; c += 32)
      out[i * (long)C + c] += gs[i * (long)C + c] * al;   // only this warp touches row i
  }
}

// out[i] = leakyrelu(acc[i] + b1[c] (+ b2[c]))
__global__ void k_bias_act(const float* __restrict__ acc, const float* __restrict__ b1,
                           const float* __restrict__ b2, float* __restrict__ out,
                           long n, int C, float slope){
  long i = blockIdx.x * (long)blockDim.x + threadIdx.x;
  long st = (long)gridDim.x * blockDim.x;
  long tot = n * (long)C;
  for (; i < tot; i += st){
    int c = (int)(i % C);
    float v = acc[i] + b1[c] + (b2 ? b2[c] : 0.f);
    out[i] = lrelu(v, slope);
  }
}

__global__ void k_final(const float* __restrict__ hp, const float* __restrict__ W,
                        const float* __restrict__ b, float* __restrict__ out, long n){
  long i = blockIdx.x * (long)blockDim.x + threadIdx.x;
  long st = (long)gridDim.x * blockDim.x;
  for (; i < n; i += st){
    float a = b[0];
    for (int c = 0; c < 32; ++c) a += hp[i * 32 + c] * W[c];
    out[i] = a;
  }
}

extern "C" void kernel_launch(void* const* d_in, const int* in_sizes, int n_in,
                              void* d_out, int out_size, void* d_ws, size_t ws_size,
                              hipStream_t stream)
{
  (void)n_in; (void)out_size; (void)ws_size;
  const float* x_b   = (const float*)d_in[0];
  const float* x_p   = (const float*)d_in[1];
  const int*   ei_bb = (const int*)d_in[2];
  const int*   ei_pp = (const int*)d_in[3];
  const int*   bpS   = (const int*)d_in[4];
  const int*   bpD   = (const int*)d_in[5];
  const float* W_in_b=(const float*)d_in[6];  const float* b_in_b=(const float*)d_in[7];
  const float* W_in_p=(const float*)d_in[8];  const float* b_in_p=(const float*)d_in[9];
  const float* W_bb1 =(const float*)d_in[10]; const float* as_bb1=(const float*)d_in[11];
  const float* ad_bb1=(const float*)d_in[12]; const float* b_bb1 =(const float*)d_in[13];
  const float* W_pp1 =(const float*)d_in[14]; const float* as_pp1=(const float*)d_in[15];
  const float* ad_pp1=(const float*)d_in[16]; const float* b_pp1 =(const float*)d_in[17];
  const float* Ws_bp1=(const float*)d_in[18]; const float* Wd_bp1=(const float*)d_in[19];
  const float* as_bp1=(const float*)d_in[20]; const float* ad_bp1=(const float*)d_in[21];
  const float* b_bp1 =(const float*)d_in[22];
  // d_in[23..26] = W_bb2/as/ad/b  -> dead (ob2 never consumed by reference output)
  const float* W_pp2 =(const float*)d_in[27]; const float* as_pp2=(const float*)d_in[28];
  const float* ad_pp2=(const float*)d_in[29]; const float* b_pp2 =(const float*)d_in[30];
  const float* Ws_bp2=(const float*)d_in[31]; const float* Wd_bp2=(const float*)d_in[32];
  const float* as_bp2=(const float*)d_in[33]; const float* ad_bp2=(const float*)d_in[34];
  const float* b_bp2 =(const float*)d_in[35];
  const float* W_out =(const float*)d_in[36]; const float* b_out =(const float*)d_in[37];
  float* out = (float*)d_out;

  const int FB = 32, FP = 34, H = 64, H2 = 32;
  const long NB  = in_sizes[0] / FB;
  const long NP  = in_sizes[1] / FP;
  const long Ebb = in_sizes[2] / 2;
  const long Epp = in_sizes[3] / 2;
  const long Ebp = in_sizes[4];
  const long NBp = (NB + 63) & ~63L;    // M padded to 64 (4 waves x 16 rows)
  const long NPp = (NP + 63) & ~63L;

  // ---- workspace bump allocator ----
  char* p = (char*)d_ws;
  auto alloc = [&](size_t bytes)->char* { char* r = p; p += (bytes + 255) & ~(size_t)255; return r; };
  float*    hb   = (float*)alloc((size_t)NB * H * 4);
  float*    hp   = (float*)alloc((size_t)NP * H * 4);
  float*    gb   = (float*)alloc((size_t)NB * H * 4);   // src-transformed feats (b side)
  float*    gp   = (float*)alloc((size_t)NP * H * 4);   // transformed feats (p side)
  float*    ob   = (float*)alloc((size_t)NB * H * 4);   // b-dst accumulator
  float*    op   = (float*)alloc((size_t)NP * H * 4);   // p-dst accumulator
  float*    ss   = (float*)alloc((size_t)NB * 4);       // src scores
  float*    sd   = (float*)alloc((size_t)NB * 4);       // dst scores
  unsigned* mx   = (unsigned*)alloc((size_t)NB * 4);    // segment max (encoded)
  float*    dn   = (float*)alloc((size_t)NB * 4);       // segment denom
  _Float16* h16b = (_Float16*)alloc((size_t)NBp * H * 2);  // padded f16 activations (b)
  _Float16* h16p = (_Float16*)alloc((size_t)NPp * H * 2);  // padded f16 activations (p)
  _Float16* w16  = (_Float16*)alloc((size_t)H * H * 2);    // transposed padded f16 weight

  auto ew = [&](long n){ long g = (n + 255) / 256; if (g > 4096) g = 4096; if (g < 1) g = 1; return (int)g; };
  auto cvtA = [&](const float* s, _Float16* d, long M, int K, long Mp, int Kp){
    k_a2h<<<ew(Mp*(long)Kp),256,0,stream>>>(s,d,M,K,Mp,Kp);
  };
  auto cvtW = [&](const float* w, int K, int N, int Kp){
    k_w2ht<<<ew((long)N*Kp),256,0,stream>>>(w,w16,K,N,Kp);
  };
  auto zf = [&](float* q, long n){ k_fill_f32<<<ew(n),256,0,stream>>>(q,0.f,n); };
  auto zu = [&](unsigned* q, long n){ k_fill_u32<<<ew(n),256,0,stream>>>(q,0u,n); };
  auto gemm = [&](const _Float16* A, const float* bias, float* C,
                  long M, long Mp, int Kp, int N, int act){
    dim3 g((unsigned)(Mp / 64)); dim3 b(32, 4);
    if (N == 64) k_gemm_wmma<4><<<g,b,0,stream>>>(A,w16,bias,C,(int)M,Kp,N,act,NEG_LR);
    else         k_gemm_wmma<2><<<g,b,0,stream>>>(A,w16,bias,C,(int)M,Kp,N,act,NEG_LR);
  };
  auto dots = [&](const float* g, const float* a, float* s, long n, int C){
    k_dot<<<ew(n),256,0,stream>>>(g,a,s,n,C);
  };
  auto wgrid = [&](long n){ long g = (n * 32 + 255) / 256; if (g > 16384) g = 16384; if (g < 1) g = 1; return (int)g; };

  // one GAT relation: project -> score -> softmax (2 passes) -> scatter
  auto gat = [&](const _Float16* hsrc, const _Float16* hdst,
                 long n_src, long n_src_p, long n_dst, long n_dst_p,
                 const float* Wsrc, const float* Wdst, bool shared,
                 const float* a_s, const float* a_d,
                 const int* src, const int* dst, long E, bool self,
                 int Cin, int Cout, float* gs, float* gd, float* outacc){
    const int Kp = (Cin + 31) & ~31;
    cvtW(Wsrc, Cin, Cout, Kp);
    gemm(hsrc, nullptr, gs, n_src, n_src_p, Kp, Cout, 0);
    if (!shared){
      cvtW(Wdst, Cin, Cout, Kp);
      gemm(hdst, nullptr, gd, n_dst, n_dst_p, Kp, Cout, 0);
    }
    const float* gdd = shared ? gs : gd;
    dots(gs,  a_s, ss, n_src, Cout);
    dots(gdd, a_d, sd, n_dst, Cout);
    zu(mx, n_dst);
    k_edge_max<<<ew(E),256,0,stream>>>(src,dst,ss,sd,mx,E);
    if (self) k_loop_max<<<ew(n_dst),256,0,stream>>>(ss,sd,mx,n_dst);
    zf(dn, n_dst);
    k_edge_sum<<<ew(E),256,0,stream>>>(src,dst,ss,sd,mx,dn,E);
    if (self) k_loop_sum<<<ew(n_dst),256,0,stream>>>(ss,sd,mx,dn,n_dst);
    k_edge_scatter<<<wgrid(E),256,0,stream>>>(src,dst,ss,sd,mx,dn,gs,outacc,E,Cout);
    if (self) k_loop_scatter<<<wgrid(n_dst),256,0,stream>>>(ss,sd,mx,dn,gs,outacc,n_dst,Cout);
  };

  // ---- input projections (fused bias + leakyrelu 0.01) ----
  cvtA(x_b, h16b, NB, FB, NBp, 32);
  cvtW(W_in_b, FB, H, 32);
  gemm(h16b, b_in_b, hb, NB, NBp, 32, H, 1);
  cvtA(x_p, h16p, NP, FP, NPp, 64);                   // K=34 zero-padded to 64
  cvtW(W_in_p, FP, H, 64);
  gemm(h16p, b_in_p, hp, NP, NPp, 64, H, 1);

  // ---- layer 1 ----
  cvtA(hb, h16b, NB, H, NBp, 64);
  cvtA(hp, h16p, NP, H, NPp, 64);
  zf(ob, NB * (long)H);
  zf(op, NP * (long)H);
  gat(h16b, h16b, NB, NBp, NB, NBp, W_bb1, W_bb1, true,  as_bb1, ad_bb1,
      ei_bb, ei_bb + Ebb, Ebb, true,  H, H, gb, gb, ob);
  gat(h16p, h16p, NP, NPp, NP, NPp, W_pp1, W_pp1, true,  as_pp1, ad_pp1,
      ei_pp, ei_pp + Epp, Epp, true,  H, H, gp, gp, op);
  gat(h16b, h16p, NB, NBp, NP, NPp, Ws_bp1, Wd_bp1, false, as_bp1, ad_bp1,
      bpS, bpD, Ebp, false, H, H, gb, gp, op);
  k_bias_act<<<ew(NB*(long)H),256,0,stream>>>(ob, b_bb1, nullptr, hb, NB, H, NEG_LR);
  k_bias_act<<<ew(NP*(long)H),256,0,stream>>>(op, b_pp1, b_bp1,  hp, NP, H, NEG_LR);

  // ---- layer 2 (bb2 branch is dead in the reference) ----
  cvtA(hb, h16b, NB, H, NBp, 64);
  cvtA(hp, h16p, NP, H, NPp, 64);
  zf(op, NP * (long)H2);
  gat(h16p, h16p, NP, NPp, NP, NPp, W_pp2, W_pp2, true,  as_pp2, ad_pp2,
      ei_pp, ei_pp + Epp, Epp, true,  H, H2, gp, gp, op);
  gat(h16b, h16p, NB, NBp, NP, NPp, Ws_bp2, Wd_bp2, false, as_bp2, ad_bp2,
      bpS, bpD, Ebp, false, H, H2, gb, gp, op);
  k_bias_act<<<ew(NP*(long)H2),256,0,stream>>>(op, b_pp2, b_bp2, hp, NP, H2, NEG_LR);

  // ---- readout ----
  k_final<<<ew(NP),256,0,stream>>>(hp, W_out, b_out, out, NP);
}
